// VRAE_LSTM_19189913878819
// MI455X (gfx1250) — compile-verified, matrix operand order checked by reading, not run on hardware
//
#include <hip/hip_runtime.h>

typedef __attribute__((ext_vector_type(16))) _Float16 v16h;
typedef __attribute__((ext_vector_type(8)))  float    v8f;
typedef __attribute__((ext_vector_type(4)))  int      v4i;

#define Bsz   512
#define Tsz   512
#define HE    512
#define Lz    64
#define HD    512
#define NG    2048     // 4*H gate columns
#define MT    16       // batch rows per workgroup
#define NTW   8        // 16-col n-tiles per wave (8*16 = 128 cols/wave)
#define GPAD  2056     // padded gate row (f16)
#define HPAD  584      // padded h row (f16), holds K up to 576
#define KCE   16       // encoder K-chunks (512/32)
#define KCD   18       // decoder K-chunks (576/32: [h|z])

union ABreg { int i[8]; v4i q[2]; v16h h; };

__device__ __forceinline__ float sigmoid_f(float v) {
  return 1.f / (1.f + __expf(-v));
}
__device__ __forceinline__ float softplus_f(float v) {
  return fmaxf(v, 0.f) + log1pf(__expf(-fabsf(v)));
}

// -----------------------------------------------------------------------------
// Pack U (rows x 2048, f32) into the exact WMMA-B operand layout, f16.
// Tile (kc, nt): 32 lanes x 8 dwords. dword d, lane l holds K pair
// k0 = (d<4 ? 2d : 16+2(d-4)) + (l>=16 ? 8 : 0), column n = nt*16 + (l&15).
// Rows >= 512 come from Wext (decoder K-extension: dec_W stacked under dec_U).
// -----------------------------------------------------------------------------
__global__ void pack_U_kernel(const float* __restrict__ U,
                              const float* __restrict__ Wext,
                              int kcCount, unsigned int* __restrict__ dst) {
  int idx = blockIdx.x * blockDim.x + threadIdx.x;
  int total = kcCount << 15;               // kc*128*32*8
  if (idx >= total) return;
  int d    = idx & 7;
  int lane = (idx >> 3) & 31;
  int nt   = (idx >> 8) & 127;
  int kc   = idx >> 15;
  int k0 = ((d < 4) ? (2 * d) : (16 + 2 * (d - 4))) + ((lane >= 16) ? 8 : 0);
  int n  = nt * 16 + (lane & 15);
  int k  = kc * 32 + k0;                   // k0 is even -> pair stays in one source
  float a = (k < 512) ? U[k * NG + n] : Wext[(k - 512) * NG + n];
  float b = (k < 512) ? U[(k + 1) * NG + n] : Wext[(k + 1 - 512) * NG + n];
  _Float16 ha = (_Float16)a, hb = (_Float16)b;
  unsigned int ua = (unsigned int)__builtin_bit_cast(unsigned short, ha);
  unsigned int ub = (unsigned int)__builtin_bit_cast(unsigned short, hb);
  dst[idx] = ua | (ub << 16);
}

// -----------------------------------------------------------------------------
// Persistent VRAE kernel: one workgroup = 16 batch rows, full T loop inside.
// -----------------------------------------------------------------------------
__global__ void __launch_bounds__(512)
vrae_kernel(const float* __restrict__ x,        // [B,T,1]
            const float* __restrict__ encW,     // [2048]
            const float* __restrict__ encb,     // [2048]
            const unsigned int* __restrict__ encUpk,
            const float* __restrict__ emuW, const float* __restrict__ emub,
            const float* __restrict__ estdW, const float* __restrict__ estdb,
            const float* __restrict__ dfsW, const float* __restrict__ dfsb,
            const unsigned int* __restrict__ decUpk,   // K=576 packed
            const float* __restrict__ decb,
            const float* __restrict__ dmuW, const float* __restrict__ dmub,
            const float* __restrict__ dstdW, const float* __restrict__ dstdb,
            float* __restrict__ out) {
  __shared__ _Float16 gatesS[MT][GPAD];   // activated gates, f16
  __shared__ _Float16 hS[MT][HPAD];       // h (cols 0..511) | z (512..575), f16
  __shared__ float    zS[MT][Lz];
  __shared__ float    xsS[MT];

  const int tid   = threadIdx.x;
  const int lane  = tid & 31;
  const int w     = tid >> 5;             // wave id 0..15 (also update-phase row)
  const int nlo   = lane & 15;
  const int mhi   = (lane >> 4) << 3;     // 0 or 8 (upper half-wave -> rows 8..15)
  const int bbase = blockIdx.x * MT;

  // zero h/z region
  for (int i = tid; i < MT * HPAD; i += 512) ((_Float16*)hS)[i] = (_Float16)0.f;

  // per-lane, per-n-tile constants (column n is fixed per lane per tile)
  float biasE[NTW], wxE[NTW], biasD[NTW];
  bool  isG[NTW];
  for (int nt = 0; nt < NTW; ++nt) {
    int n = w * 128 + nt * 16 + nlo;
    biasE[nt] = encb[n];
    wxE[nt]   = encW[n];
    biasD[nt] = decb[n];
    isG[nt]   = ((n >> 9) == 2);          // gate region 2 = g -> tanh
  }

  float c[16];                            // cell state, registers (fixed mapping)
  #pragma unroll
  for (int p = 0; p < 16; ++p) c[p] = 0.f;

  const v8f vzero = {0.f, 0.f, 0.f, 0.f, 0.f, 0.f, 0.f, 0.f};
  __syncthreads();

  // ============================ encoder =====================================
  for (int t = 0; t < Tsz; ++t) {
    if (tid < MT) xsS[tid] = x[(bbase + tid) * Tsz + t];
    __syncthreads();

    v8f acc[NTW];
    #pragma unroll
    for (int nt = 0; nt < NTW; ++nt) acc[nt] = vzero;

    for (int kc = 0; kc < KCE; ++kc) {
      ABreg A;
      #pragma unroll
      for (int d = 0; d < 8; ++d) {
        int k0 = ((d < 4) ? (2 * d) : (16 + 2 * (d - 4))) + ((lane >= 16) ? 8 : 0);
        A.i[d] = *(const int*)&hS[nlo][kc * 32 + k0];
      }
      const unsigned int* bp = encUpk + (((kc * 128) + w * 8) * 32 + lane) * 8;
      #pragma unroll
      for (int nt = 0; nt < NTW; ++nt) {
        ABreg Bm;
        Bm.q[0] = *(const v4i*)(bp);
        Bm.q[1] = *(const v4i*)(bp + 4);
        bp += 32 * 8;
        acc[nt] = __builtin_amdgcn_wmma_f32_16x16x32_f16(
            false, A.h, false, Bm.h, (short)0, acc[nt], false, false);
      }
    }
    // bias + x*W, activate per gate region, stage to LDS (f16)
    #pragma unroll
    for (int nt = 0; nt < NTW; ++nt) {
      int n = w * 128 + nt * 16 + nlo;
      #pragma unroll
      for (int r = 0; r < 8; ++r) {
        int m = r + mhi;
        float v = acc[nt][r] + biasE[nt] + xsS[m] * wxE[nt];
        v = isG[nt] ? tanhf(v) : sigmoid_f(v);
        gatesS[m][n] = (_Float16)v;
      }
    }
    __syncthreads();
    // elementwise LSTM update: wave w owns row w, lane owns 16 hidden units
    #pragma unroll
    for (int p = 0; p < 16; ++p) {
      int j = lane * 16 + p;
      float iv = (float)gatesS[w][j];
      float fv = (float)gatesS[w][512 + j];
      float gv = (float)gatesS[w][1024 + j];
      float ov = (float)gatesS[w][1536 + j];
      c[p] = fv * c[p] + iv * gv;
      hS[w][j] = (_Float16)(ov * tanhf(c[p]));
    }
    __syncthreads();
  }

  // ===================== latent heads: mu_enc / sigma_enc ===================
  for (int l = 0; l < Lz; ++l) {
    float pm = 0.f, ps = 0.f;
    #pragma unroll
    for (int p = 0; p < 16; ++p) {
      int j = lane + 32 * p;
      float hv = (float)hS[w][j];
      pm += hv * emuW[j * Lz + l];
      ps += hv * estdW[j * Lz + l];
    }
    #pragma unroll
    for (int off = 16; off; off >>= 1) {
      pm += __shfl_xor(pm, off, 32);
      ps += __shfl_xor(ps, off, 32);
    }
    if (lane == 0) {
      float mu = pm + emub[l];
      float sg = ps + estdb[l];
      zS[w][l] = mu;                       // z = mu (reparam=False)
      int bg = bbase + w;
      out[2 * Bsz * Tsz + bg * Lz + l] = mu;
      out[2 * Bsz * Tsz + Bsz * Lz + bg * Lz + l] = softplus_f(sg);
    }
  }
  __syncthreads();

  // =============== decoder init: hd = tanh(z@dfs_W + b), c = 0 =============
  #pragma unroll
  for (int p = 0; p < 16; ++p) {
    int j = lane * 16 + p;
    float a = dfsb[j];
    for (int l = 0; l < Lz; ++l) a += zS[w][l] * dfsW[l * HD + j];
    hS[w][j] = (_Float16)tanhf(a);
    c[p] = 0.f;
  }
  for (int l = lane; l < Lz; l += 32) hS[w][512 + l] = (_Float16)zS[w][l];
  __syncthreads();

  // ============================ decoder =====================================
  // gates = [h | z] @ [dec_U ; dec_W] + dec_b   (K = 576)
  for (int t = 0; t < Tsz; ++t) {
    v8f acc[NTW];
    #pragma unroll
    for (int nt = 0; nt < NTW; ++nt) acc[nt] = vzero;

    for (int kc = 0; kc < KCD; ++kc) {
      ABreg A;
      #pragma unroll
      for (int d = 0; d < 8; ++d) {
        int k0 = ((d < 4) ? (2 * d) : (16 + 2 * (d - 4))) + ((lane >= 16) ? 8 : 0);
        A.i[d] = *(const int*)&hS[nlo][kc * 32 + k0];
      }
      const unsigned int* bp = decUpk + (((kc * 128) + w * 8) * 32 + lane) * 8;
      #pragma unroll
      for (int nt = 0; nt < NTW; ++nt) {
        ABreg Bm;
        Bm.q[0] = *(const v4i*)(bp);
        Bm.q[1] = *(const v4i*)(bp + 4);
        bp += 32 * 8;
        acc[nt] = __builtin_amdgcn_wmma_f32_16x16x32_f16(
            false, A.h, false, Bm.h, (short)0, acc[nt], false, false);
      }
    }
    #pragma unroll
    for (int nt = 0; nt < NTW; ++nt) {
      int n = w * 128 + nt * 16 + nlo;
      #pragma unroll
      for (int r = 0; r < 8; ++r) {
        int m = r + mhi;
        float v = acc[nt][r] + biasD[nt];
        v = isG[nt] ? tanhf(v) : sigmoid_f(v);
        gatesS[m][n] = (_Float16)v;
      }
    }
    __syncthreads();

    float pm = 0.f, ps = 0.f;
    #pragma unroll
    for (int p = 0; p < 16; ++p) {
      int j = lane * 16 + p;
      float iv = (float)gatesS[w][j];
      float fv = (float)gatesS[w][512 + j];
      float gv = (float)gatesS[w][1024 + j];
      float ov = (float)gatesS[w][1536 + j];
      c[p] = fv * c[p] + iv * gv;
      float hv = ov * tanhf(c[p]);
      hS[w][j] = (_Float16)hv;
      pm += hv * dmuW[j];
      ps += hv * dstdW[j];
    }
    #pragma unroll
    for (int off = 16; off; off >>= 1) {
      pm += __shfl_xor(pm, off, 32);
      ps += __shfl_xor(ps, off, 32);
    }
    if (lane == 0) {
      int bg = bbase + w;
      out[bg * Tsz + t] = pm + dmub[0];
      out[Bsz * Tsz + bg * Tsz + t] = softplus_f(ps + dstdb[0]);
    }
    __syncthreads();
  }
}

// -----------------------------------------------------------------------------
extern "C" void kernel_launch(void* const* d_in, const int* in_sizes, int n_in,
                              void* d_out, int out_size, void* d_ws, size_t ws_size,
                              hipStream_t stream) {
  const float* x     = (const float*)d_in[0];
  const float* encW  = (const float*)d_in[1];
  const float* encU  = (const float*)d_in[2];
  const float* encb  = (const float*)d_in[3];
  const float* emuW  = (const float*)d_in[4];
  const float* emub  = (const float*)d_in[5];
  const float* estdW = (const float*)d_in[6];
  const float* estdb = (const float*)d_in[7];
  const float* dfsW  = (const float*)d_in[8];
  const float* dfsb  = (const float*)d_in[9];
  const float* decW  = (const float*)d_in[10];
  const float* decU  = (const float*)d_in[11];
  const float* decb  = (const float*)d_in[12];
  const float* dmuW  = (const float*)d_in[13];
  const float* dmub  = (const float*)d_in[14];
  const float* dstdW = (const float*)d_in[15];
  const float* dstdb = (const float*)d_in[16];

  unsigned int* encUpk = (unsigned int*)d_ws;          // 16*32768 dwords = 2 MB
  unsigned int* decUpk = encUpk + (KCE << 15);         // 18*32768 dwords = 2.25 MB

  int totE = KCE << 15;
  int totD = KCD << 15;
  pack_U_kernel<<<(totE + 255) / 256, 256, 0, stream>>>(encU, encU, KCE, encUpk);
  pack_U_kernel<<<(totD + 255) / 256, 256, 0, stream>>>(decU, decW, KCD, decUpk);

  vrae_kernel<<<Bsz / MT, 512, 0, stream>>>(
      x, encW, encb, encUpk, emuW, emub, estdW, estdb,
      dfsW, dfsb, decUpk, decb, dmuW, dmub, dstdW, dstdb, (float*)d_out);
}
